// TTTAttention_86981677679350
// MI455X (gfx1250) — matrix operand order repaired and checked by reading, not compiled, not run on hardware
//
#include <hip/hip_runtime.h>

// ---------------------------------------------------------------------------
// Types
// ---------------------------------------------------------------------------
typedef __attribute__((ext_vector_type(16))) __bf16        v16bf;
typedef __attribute__((ext_vector_type(8)))  float         v8f;
typedef __attribute__((ext_vector_type(4)))  unsigned int  u32x4;
typedef __attribute__((ext_vector_type(8)))  int           i32x8;
typedef __attribute__((ext_vector_type(4)))  int           i32x4;

struct alignas(16) U128 { unsigned int x, y, z, w; };

// ---------------------------------------------------------------------------
// Problem constants (from the reference)
// ---------------------------------------------------------------------------
#define N_TOK 10992            // B*SN
#define NPAD  11008            // padded: multiple of 128 (M) and 64 (K)
#define DIM_  1024
#define H3_   3072
#define HID_  4096
#define LR_   0.1f
#define NS_A  3.4445f
#define NS_B  (-4.7750f)
#define NS_C  2.0315f

__device__ __forceinline__ unsigned short f2b(float f) {
  unsigned int u = __float_as_uint(f);
  u += 0x7FFFu + ((u >> 16) & 1u);       // round-to-nearest-even bf16
  return (unsigned short)(u >> 16);
}
__device__ __forceinline__ unsigned short toB(float v) { return f2b(v); }
__device__ __forceinline__ unsigned short toB(unsigned short v) { return v; }

// ---------------------------------------------------------------------------
// Tensor Data Mover path detection
// ---------------------------------------------------------------------------
#ifdef __HIP_DEVICE_COMPILE__
#  if __has_builtin(__builtin_amdgcn_tensor_load_to_lds) && \
      __has_builtin(__builtin_amdgcn_s_wait_tensorcnt)
#    define USE_TDM 1
#  endif
#endif
#ifndef USE_TDM
#  define USE_TDM 0
#endif

#if USE_TDM
// Build D# per CDNA5 ISA and issue TENSOR_LOAD_TO_LDS.
// 2D tile of bf16 (data_size=1 => 2 bytes): tile_dim0 x tile_dim1,
// row stride = strideElems (elements).
__device__ __forceinline__ void tdm_load_tile(const void* gptr, unsigned int ldsOff,
                                              unsigned int tile0, unsigned int tile1,
                                              unsigned long long strideElems) {
  unsigned long long ga = (unsigned long long)(size_t)gptr;
  u32x4 g0;
  g0[0] = 1u;                                              // count=1, user mode
  g0[1] = ldsOff;                                          // lds_addr (bytes)
  g0[2] = (unsigned int)(ga & 0xFFFFFFFFu);                // global_addr[31:0]
  g0[3] = (unsigned int)((ga >> 32) & 0x01FFFFFFu) | (2u << 30); // [56:32] | type=2
  i32x8 g1;
  g1[0] = (int)(1u << 16);                                 // data_size=1 (2 bytes)
  g1[1] = (int)((strideElems & 0xFFFFull) << 16);          // tensor_dim0[15:0]
  unsigned int td1 = 1u << 20;                             // tensor_dim1 (large)
  g1[2] = (int)(((strideElems >> 16) & 0xFFFFull) |
                ((unsigned long long)(td1 & 0xFFFFu) << 16));
  g1[3] = (int)(((td1 >> 16) & 0xFFFFu) | (tile0 << 16));  // tile_dim0
  g1[4] = (int)(tile1 & 0xFFFFu);                          // tile_dim1 (tile_dim2=0)
  g1[5] = (int)(strideElems & 0xFFFFFFFFull);              // tensor_dim0_stride lo
  g1[6] = (int)((strideElems >> 32) & 0xFFFFull);          // stride hi (dim1_stride=0)
  g1[7] = 0;
  i32x4 z4; z4[0] = z4[1] = z4[2] = z4[3] = 0;
#if __clang_major__ >= 23
  i32x8 z8; for (int i = 0; i < 8; ++i) z8[i] = 0;
  __builtin_amdgcn_tensor_load_to_lds(g0, g1, z4, z4, z8, 0);
#else
  __builtin_amdgcn_tensor_load_to_lds(g0, g1, z4, z4, 0);
#endif
}
#endif

// ---------------------------------------------------------------------------
// WMMA GEMM:  C[M,N] (f32) = A[M,K] (bf16 row-major) @ B, where the second
// operand is supplied COLUMN-MAJOR as BT[N,K] (bf16 row-major, ldbt >= K).
// Block tile 128x128, BK=64, 8 waves, wave tile 64x32 (4x2 WMMA tiles).
// Double-buffered LDS (2 x 32KB); TDM loads tile i+1 while computing tile i.
// M,N multiples of 128, K multiple of 64. Stores only rows < Mstore.
// ---------------------------------------------------------------------------
__global__ __launch_bounds__(256) void gemm_bf16(
    const unsigned short* __restrict__ A, const unsigned short* __restrict__ BT,
    float* __restrict__ C, int K, int lda, int ldbt, int ldc, int Mstore,
    const float* __restrict__ bias) {
  extern __shared__ unsigned short smem[];   // buf b: A @ b*32768B, B @ +16384B

  const int tid  = threadIdx.x;
  const int lane = tid & 31;
  const int wave = tid >> 5;
  const int wm   = wave >> 2;                // 0..1 -> 64 rows
  const int wn   = wave & 3;                 // 0..3 -> 32 cols
  const long bm0 = (long)blockIdx.y * 128;
  const long bn0 = (long)blockIdx.x * 128;
  const int nIter = K >> 6;

  v8f acc[4][2];
  for (int i = 0; i < 4; ++i)
    for (int j = 0; j < 2; ++j)
      for (int e = 0; e < 8; ++e) acc[i][j][e] = 0.0f;

#if USE_TDM
  if (tid < 32) {                            // prologue: fetch tile 0 into buf 0
    tdm_load_tile(A + bm0 * lda, 0u, 64u, 128u, (unsigned long long)lda);
    tdm_load_tile(BT + bn0 * ldbt, 16384u, 64u, 128u, (unsigned long long)ldbt);
    __builtin_amdgcn_s_wait_tensorcnt(0);
  }
  __syncthreads();
#endif

  for (int i = 0; i < nIter; ++i) {
    const int b = i & 1;
#if USE_TDM
    if (tid < 32 && (i + 1) < nIter) {       // prefetch next tile into other buf
      int kn = (i + 1) << 6;
      unsigned int lds = (unsigned int)(1 - b) * 32768u;
      tdm_load_tile(A + bm0 * lda + kn, lds, 64u, 128u, (unsigned long long)lda);
      tdm_load_tile(BT + bn0 * ldbt + kn, lds + 16384u, 64u, 128u,
                    (unsigned long long)ldbt);
    }
#else
    {
      int k0 = i << 6;
      unsigned short* AsW = smem + b * 16384;
      unsigned short* BsW = AsW + 8192;
      for (int it = tid; it < 1024; it += 256) {   // A tile 128x64
        int r = it >> 3, c = (it & 7) * 8;
        *(U128*)&AsW[r * 64 + c] = *(const U128*)&A[(bm0 + r) * lda + k0 + c];
      }
      for (int it = tid; it < 1024; it += 256) {   // BT tile 128x64
        int r = it >> 3, c = (it & 7) * 8;
        *(U128*)&BsW[r * 64 + c] = *(const U128*)&BT[(bn0 + r) * ldbt + k0 + c];
      }
      __syncthreads();
    }
#endif
    const unsigned short* As = smem + b * 16384;
    const unsigned short* Bs = As + 8192;
    const int mr = lane & 15;
    const int kh = lane >> 4;                // A: K-half; B: K-group
#pragma unroll
    for (int ks = 0; ks < 64; ks += 32) {
      v16bf af[4];
#pragma unroll
      for (int t = 0; t < 4; ++t) {          // A frag: K = ks + {kh*8+0..7, 16+kh*8+0..7}
        int m = wm * 64 + t * 16 + mr;
        union UA { U128 q[2]; v16bf v; } ua;
        ua.q[0] = *(const U128*)&As[m * 64 + ks + kh * 8];
        ua.q[1] = *(const U128*)&As[m * 64 + ks + 16 + kh * 8];
        af[t] = ua.v;
      }
      v16bf bfr[2];
#pragma unroll
      for (int t = 0; t < 2; ++t) {          // B frag: K = ks + {kh*8+0..7, 16+kh*8+0..7}
        int n = wn * 32 + t * 16 + mr;
        union UB { U128 q[2]; v16bf v; } ub;
        ub.q[0] = *(const U128*)&Bs[n * 64 + ks + kh * 8];
        ub.q[1] = *(const U128*)&Bs[n * 64 + ks + 16 + kh * 8];
        bfr[t] = ub.v;
      }
#pragma unroll
      for (int i2 = 0; i2 < 4; ++i2)
#pragma unroll
        for (int j2 = 0; j2 < 2; ++j2)
          acc[i2][j2] = __builtin_amdgcn_wmma_f32_16x16x32_bf16(
              false, af[i2], false, bfr[j2], (short)0, acc[i2][j2], false, false);
    }
    __syncthreads();
#if USE_TDM
    if (tid < 32) __builtin_amdgcn_s_wait_tensorcnt(0);
    __syncthreads();
#endif
  }

#pragma unroll
  for (int i = 0; i < 4; ++i)
#pragma unroll
    for (int j = 0; j < 2; ++j) {
      long gn = bn0 + wn * 32 + j * 16 + (lane & 15);
      float bv = bias ? bias[gn] : 0.0f;
      long gm0 = bm0 + wm * 64 + i * 16 + (lane >> 4) * 8;
#pragma unroll
      for (int e = 0; e < 8; ++e) {
        long gm = gm0 + e;
        if (gm < Mstore) C[gm * ldc + gn] = acc[i][j][e] + bv;
      }
    }
}

// ---------------------------------------------------------------------------
// Elementwise / helper kernels
// ---------------------------------------------------------------------------
__global__ void cvt_k(const float* __restrict__ in, unsigned short* __restrict__ out,
                      int nsrc, int ntot) {
  int i = blockIdx.x * 256 + threadIdx.x;
  if (i >= ntot) return;
  out[i] = (i < nsrc) ? f2b(in[i]) : (unsigned short)0;
}

__global__ void copycvt_k(const float* __restrict__ src, float* __restrict__ dst,
                          unsigned short* __restrict__ dstb, int n) {
  int i = blockIdx.x * 256 + threadIdx.x;
  if (i >= n) return;
  float v = src[i]; dst[i] = v; dstb[i] = f2b(v);
}

template <typename T>
__global__ __launch_bounds__(256) void transpose_cvt(const T* __restrict__ in,
                                                     unsigned short* __restrict__ out,
                                                     int R, int C) {
  __shared__ unsigned short tile[32][33];
  int x = blockIdx.x * 32 + threadIdx.x;
  for (int j = threadIdx.y; j < 32; j += 8) {
    int y = blockIdx.y * 32 + j;
    unsigned short v = 0;
    if (y < R && x < C) v = toB(in[(long)y * C + x]);
    tile[j][threadIdx.x] = v;
  }
  __syncthreads();
  int ox = blockIdx.y * 32 + threadIdx.x;     // new col = old row
  for (int j = threadIdx.y; j < 32; j += 8) {
    int oy = blockIdx.x * 32 + j;             // new row = old col
    if (oy < C && ox < R) out[(long)oy * R + ox] = tile[threadIdx.x][j];
  }
}

// QKV bias + split + per-head L2 norm + affine. One block per token row.
__global__ __launch_bounds__(256) void split_norm_k(
    const float* __restrict__ qkv, const float* __restrict__ qkvb,
    const float* __restrict__ qs, const float* __restrict__ qb,
    const float* __restrict__ ks, const float* __restrict__ kb,
    unsigned short* __restrict__ q_b, unsigned short* __restrict__ k_b,
    float* __restrict__ v_f, int nreal) {
  int row = blockIdx.x;
  int t = threadIdx.x;
  int e0 = t * 4;                         // 4 contiguous elems, within one head
  long qbase = (long)row * DIM_ + e0;
  if (row >= nreal) {
    for (int j = 0; j < 4; ++j) { q_b[qbase + j] = 0; k_b[qbase + j] = 0; }
    return;
  }
  const float* rq = qkv + (long)row * H3_;
  float qv[4], kv[4], vv[4], ssq = 0.f, ssk = 0.f;
  for (int j = 0; j < 4; ++j) {
    qv[j] = rq[e0 + j] + qkvb[e0 + j];
    kv[j] = rq[DIM_ + e0 + j] + qkvb[DIM_ + e0 + j];
    vv[j] = rq[2 * DIM_ + e0 + j] + qkvb[2 * DIM_ + e0 + j];
    ssq += qv[j] * qv[j];
    ssk += kv[j] * kv[j];
  }
  for (int m = 1; m < 16; m <<= 1) {      // head = 16 lanes (wave32: 2 heads/wave)
    ssq += __shfl_xor(ssq, m, 16);
    ssk += __shfl_xor(ssk, m, 16);
  }
  float iq = 1.0f / (sqrtf(ssq) + 1e-6f);
  float ik = 1.0f / (sqrtf(ssk) + 1e-6f);
  int hp = e0 & 63;                       // position within head
  for (int j = 0; j < 4; ++j) {
    q_b[qbase + j] = f2b(qv[j] * iq * qs[hp + j] + qb[hp + j]);
    k_b[qbase + j] = f2b(kv[j] * ik * ks[hp + j] + kb[hp + j]);
    v_f[qbase + j] = vv[j];
  }
}

__global__ void swmul_k(const float* __restrict__ g, const float* __restrict__ u,
                        unsigned short* __restrict__ hb, int n) {
  int i = blockIdx.x * 256 + threadIdx.x;
  if (i >= n) return;
  float gg = g[i];
  float s = 1.0f / (1.0f + __expf(-gg));
  hb[i] = f2b(gg * s * u[i]);
}

__global__ void mse_k(const float* __restrict__ pred, const float* __restrict__ v,
                      unsigned short* __restrict__ dp, int nreal, int ntot, float c0) {
  int i = blockIdx.x * 256 + threadIdx.x;
  if (i >= ntot) return;
  float vv = (i < nreal) ? v[i] : 0.0f;
  dp[i] = f2b(c0 * (pred[i] - vv));
}

__global__ void backact_k(const float* __restrict__ g, const float* __restrict__ u,
                          const float* __restrict__ dh,
                          unsigned short* __restrict__ dgb,
                          unsigned short* __restrict__ dub, int n) {
  int i = blockIdx.x * 256 + threadIdx.x;
  if (i >= n) return;
  float gg = g[i], uu = u[i], dd = dh[i];
  float s = 1.0f / (1.0f + __expf(-gg));
  dub[i] = f2b(dd * gg * s);                          // d/du: swish(g)
  float dsw = s * (1.0f + gg * (1.0f - s));           // swish'(g)
  dgb[i] = f2b(dd * uu * dsw);
}

__global__ __launch_bounds__(256) void fro_part(const float* __restrict__ x,
                                                float* __restrict__ part, int n) {
  __shared__ float sm[256];
  float s = 0.0f;
  for (int i = blockIdx.x * 256 + threadIdx.x; i < n; i += 512 * 256) {
    float v = x[i]; s += v * v;
  }
  sm[threadIdx.x] = s;
  __syncthreads();
  for (int off = 128; off > 0; off >>= 1) {
    if ((int)threadIdx.x < off) sm[threadIdx.x] += sm[threadIdx.x + off];
    __syncthreads();
  }
  if (threadIdx.x == 0) part[blockIdx.x] = sm[0];
}

__global__ __launch_bounds__(256) void fro_fin(const float* __restrict__ part,
                                               float* __restrict__ sc) {
  __shared__ float sm[256];
  float s = 0.0f;
  for (int i = threadIdx.x; i < 512; i += 256) s += part[i];
  sm[threadIdx.x] = s;
  __syncthreads();
  for (int off = 128; off > 0; off >>= 1) {
    if ((int)threadIdx.x < off) sm[threadIdx.x] += sm[threadIdx.x + off];
    __syncthreads();
  }
  if (threadIdx.x == 0) sc[0] = 1.0f / (sqrtf(sm[0]) + 1e-7f);
}

// X[DIM_,HID_] = scale * (trans ? G^T : G)   (G is [HID_,DIM_] when trans)
__global__ void scale_to_k(const float* __restrict__ g, const float* __restrict__ sc,
                           float* __restrict__ X, int R, int Cc, int trans, int n) {
  int i = blockIdx.x * 256 + threadIdx.x;
  if (i >= n) return;
  int r = i / Cc, c = i - r * Cc;
  float v = trans ? g[(long)c * R + r] : g[i];
  X[i] = v * sc[0];
}

__global__ void comb_k(const float* __restrict__ A, const float* __restrict__ A2,
                       unsigned short* __restrict__ outb, int n) {
  int i = blockIdx.x * 256 + threadIdx.x;
  if (i >= n) return;
  outb[i] = f2b(NS_B * A[i] + NS_C * A2[i]);
}

__global__ void axpy_k(float* __restrict__ X, const float* __restrict__ BX, int n) {
  int i = blockIdx.x * 256 + threadIdx.x;
  if (i >= n) return;
  X[i] = NS_A * X[i] + BX[i];
}

// p[R,Cc] -= LR * (trans ? X^T : X); refresh bf16 mirror.
__global__ void muon_k(float* __restrict__ p, unsigned short* __restrict__ pb,
                       const float* __restrict__ X, int R, int Cc, int trans, int n) {
  int i = blockIdx.x * 256 + threadIdx.x;
  if (i >= n) return;
  int r = i / Cc, c = i - r * Cc;
  float xv = trans ? X[(long)c * R + r] : X[i];
  float nv = p[i] - LR_ * xv;
  p[i] = nv;
  pb[i] = f2b(nv);
}

// ---------------------------------------------------------------------------
// Host orchestration
// ---------------------------------------------------------------------------
static inline void gemm(hipStream_t s, const unsigned short* A, const unsigned short* BT,
                        float* C, int M, int N, int K, int lda, int ldbt, int ldc,
                        int Mstore, const float* bias) {
  dim3 grid(N / 128, M / 128);
  gemm_bf16<<<grid, dim3(256), 65536, s>>>(A, BT, C, K, lda, ldbt, ldc, Mstore, bias);
}

extern "C" void kernel_launch(void* const* d_in, const int* in_sizes, int n_in,
                              void* d_out, int out_size, void* d_ws, size_t ws_size,
                              hipStream_t stream) {
  (void)in_sizes; (void)n_in; (void)out_size; (void)ws_size;
  const float* x         = (const float*)d_in[0];
  const float* qkv_w     = (const float*)d_in[1];
  const float* qkv_bias  = (const float*)d_in[2];
  const float* qn_s      = (const float*)d_in[3];
  const float* qn_bi     = (const float*)d_in[4];
  const float* kn_s      = (const float*)d_in[5];
  const float* kn_bi     = (const float*)d_in[6];
  const float* ttt_gate  = (const float*)d_in[7];
  const float* ttt_val   = (const float*)d_in[8];
  const float* ttt_proj  = (const float*)d_in[9];
  const float* proj_w    = (const float*)d_in[10];
  const float* proj_bias = (const float*)d_in[11];
  float* out = (float*)d_out;

  char* base = (char*)d_ws;
  size_t cur = 0;
  auto alloc = [&](size_t bytes) -> void* {
    void* p = base + cur;
    cur += (bytes + 255) & ~(size_t)255;
    return p;
  };
  typedef unsigned short us;

  us*    xb      = (us*)   alloc((size_t)NPAD * DIM_ * 2);
  us*    qkvwT   = (us*)   alloc((size_t)H3_ * DIM_ * 2);   // qkv_w^T  [3072,1024]
  us*    projwT  = (us*)   alloc((size_t)DIM_ * DIM_ * 2);  // proj_w^T [1024,1024]
  float* dhBuf   = (float*)alloc((size_t)NPAD * HID_ * 4);  // overlaid with qkv
  float* qkvBuf  = dhBuf;                                   // NPAD*H3_*4 <= NPAD*HID_*4
  us*    q_b     = (us*)   alloc((size_t)NPAD * DIM_ * 2);
  us*    k_b     = (us*)   alloc((size_t)NPAD * DIM_ * 2);
  us*    k_bT    = (us*)   alloc((size_t)DIM_ * NPAD * 2);
  float* v_f     = (float*)alloc((size_t)N_TOK * DIM_ * 4);
  float* gBuf    = (float*)alloc((size_t)NPAD * HID_ * 4);
  float* uBuf    = (float*)alloc((size_t)NPAD * HID_ * 4);
  us*    h_b     = (us*)   alloc((size_t)NPAD * HID_ * 2);
  us*    hT_b    = (us*)   alloc((size_t)HID_ * NPAD * 2);
  float* pred    = (float*)alloc((size_t)NPAD * DIM_ * 4);
  us*    dpred_b = (us*)   alloc((size_t)NPAD * DIM_ * 2);
  us*    dpredT  = (us*)   alloc((size_t)DIM_ * NPAD * 2);  // dpred^T [1024,NPAD]
  us*    dgT     = (us*)   alloc((size_t)HID_ * NPAD * 2);  // dg^T [4096,NPAD]
  us*    duT     = (us*)   alloc((size_t)HID_ * NPAD * 2);  // du^T [4096,NPAD]
  float* gate    = (float*)alloc((size_t)DIM_ * HID_ * 4);
  float* val     = (float*)alloc((size_t)DIM_ * HID_ * 4);
  float* proj    = (float*)alloc((size_t)HID_ * DIM_ * 4);
  us*    gate_b  = (us*)   alloc((size_t)DIM_ * HID_ * 2);  // row-major mirrors
  us*    val_b   = (us*)   alloc((size_t)DIM_ * HID_ * 2);
  us*    proj_tb = (us*)   alloc((size_t)HID_ * DIM_ * 2);  // theta proj bf16 [4096,1024]
  us*    gate_bT = (us*)   alloc((size_t)HID_ * DIM_ * 2);  // gate^T [4096,1024]
  us*    val_bT  = (us*)   alloc((size_t)HID_ * DIM_ * 2);  // val^T  [4096,1024]
  us*    projT_b = (us*)   alloc((size_t)DIM_ * HID_ * 2);  // proj^T [1024,4096]
  float* ggate   = (float*)alloc((size_t)DIM_ * HID_ * 4);
  float* gval    = (float*)alloc((size_t)DIM_ * HID_ * 4);
  float* gproj   = (float*)alloc((size_t)HID_ * DIM_ * 4);
  float* Xm      = (float*)alloc((size_t)DIM_ * HID_ * 4);
  us*    Xb      = (us*)   alloc((size_t)DIM_ * HID_ * 2);
  us*    XbT     = (us*)   alloc((size_t)HID_ * DIM_ * 2);
  float* Amat    = (float*)alloc((size_t)DIM_ * DIM_ * 4);
  us*    Ab      = (us*)   alloc((size_t)DIM_ * DIM_ * 2);
  float* A2m     = (float*)alloc((size_t)DIM_ * DIM_ * 4);
  us*    Bb      = (us*)   alloc((size_t)DIM_ * DIM_ * 2);
  float* BX      = (float*)alloc((size_t)DIM_ * HID_ * 4);
  float* parts   = (float*)alloc(512 * 4);
  float* scaleS  = (float*)alloc(4);
  // dg/du row-major are only inputs to the transposes; alias dead h/hT buffers.
  us*    dg_b    = h_b;                                     // [NPAD,HID] (h dead by then)
  us*    du_b    = hT_b;                                    // same element count

  auto ew = [](int n) { return dim3((unsigned)((n + 255) / 256)); };
  const dim3 tb(32, 8);

  // ---- setup: bf16 mirrors (transposed B operands) + QKV + split/norm ----
  cvt_k<<<ew(NPAD * DIM_), 256, 0, stream>>>(x, xb, N_TOK * DIM_, NPAD * DIM_);
  transpose_cvt<float><<<dim3(H3_ / 32, DIM_ / 32), tb, 0, stream>>>(qkv_w, qkvwT, DIM_, H3_);
  transpose_cvt<float><<<dim3(DIM_ / 32, DIM_ / 32), tb, 0, stream>>>(proj_w, projwT, DIM_, DIM_);
  copycvt_k<<<ew(DIM_ * HID_), 256, 0, stream>>>(ttt_gate, gate, gate_b, DIM_ * HID_);
  copycvt_k<<<ew(DIM_ * HID_), 256, 0, stream>>>(ttt_val, val, val_b, DIM_ * HID_);
  copycvt_k<<<ew(HID_ * DIM_), 256, 0, stream>>>(ttt_proj, proj, proj_tb, HID_ * DIM_);
  transpose_cvt<float><<<dim3(HID_ / 32, DIM_ / 32), tb, 0, stream>>>(gate, gate_bT, DIM_, HID_);
  transpose_cvt<float><<<dim3(HID_ / 32, DIM_ / 32), tb, 0, stream>>>(val, val_bT, DIM_, HID_);

  gemm(stream, xb, qkvwT, qkvBuf, NPAD, H3_, DIM_, DIM_, DIM_, H3_, NPAD, nullptr);
  split_norm_k<<<NPAD, 256, 0, stream>>>(qkvBuf, qkv_bias, qn_s, qn_bi, kn_s, kn_bi,
                                         q_b, k_b, v_f, N_TOK);
  transpose_cvt<us><<<dim3(DIM_ / 32, NPAD / 32), tb, 0, stream>>>(k_b, k_bT, NPAD, DIM_);

  const float c0 = 2.0f / ((float)N_TOK * (float)DIM_);

  // ---- Muon update with Newton-Schulz (X is always [DIM_,HID_]) ----
  auto ns_muon = [&](float* P, us* Pb, float* G, int R, int Cc, int trans) {
    fro_part<<<512, 256, 0, stream>>>(G, parts, R * Cc);
    fro_fin<<<1, 256, 0, stream>>>(parts, scaleS);
    scale_to_k<<<ew(DIM_ * HID_), 256, 0, stream>>>(G, scaleS, Xm, DIM_, HID_, trans,
                                                    DIM_ * HID_);
    for (int it = 0; it < 5; ++it) {
      cvt_k<<<ew(DIM_ * HID_), 256, 0, stream>>>(Xm, Xb, DIM_ * HID_, DIM_ * HID_);
      transpose_cvt<float><<<dim3(HID_ / 32, DIM_ / 32), tb, 0, stream>>>(Xm, XbT, DIM_, HID_);
      // A = X @ X^T : second operand X^T supplied column-major == Xb itself
      gemm(stream, Xb, Xb, Amat, DIM_, DIM_, HID_, HID_, HID_, DIM_, DIM_, nullptr);
      cvt_k<<<ew(DIM_ * DIM_), 256, 0, stream>>>(Amat, Ab, DIM_ * DIM_, DIM_ * DIM_);
      // A2 = A @ A : A symmetric -> column-major form of A is A itself
      gemm(stream, Ab, Ab, A2m, DIM_, DIM_, DIM_, DIM_, DIM_, DIM_, DIM_, nullptr);
      comb_k<<<ew(DIM_ * DIM_), 256, 0, stream>>>(Amat, A2m, Bb, DIM_ * DIM_);
      // BX = B @ X : X column-major == XbT
      gemm(stream, Bb, XbT, BX, DIM_, HID_, DIM_, DIM_, DIM_, HID_, DIM_, nullptr);
      axpy_k<<<ew(DIM_ * HID_), 256, 0, stream>>>(Xm, BX, DIM_ * HID_);
    }
    muon_k<<<ew(R * Cc), 256, 0, stream>>>(P, Pb, Xm, R, Cc, trans, R * Cc);
  };

  // ---- 2 TTT steps ----
  for (int step = 0; step < 2; ++step) {
    gemm(stream, k_b, gate_bT, gBuf, NPAD, HID_, DIM_, DIM_, DIM_, HID_, NPAD, nullptr);
    gemm(stream, k_b, val_bT, uBuf, NPAD, HID_, DIM_, DIM_, DIM_, HID_, NPAD, nullptr);
    swmul_k<<<ew(NPAD * HID_), 256, 0, stream>>>(gBuf, uBuf, h_b, NPAD * HID_);
    transpose_cvt<us><<<dim3(HID_ / 32, NPAD / 32), tb, 0, stream>>>(h_b, hT_b, NPAD, HID_);
    transpose_cvt<float><<<dim3(DIM_ / 32, HID_ / 32), tb, 0, stream>>>(proj, projT_b,
                                                                        HID_, DIM_);
    // pred = h @ proj : proj column-major == projT_b
    gemm(stream, h_b, projT_b, pred, NPAD, DIM_, HID_, HID_, HID_, DIM_, NPAD, nullptr);
    mse_k<<<ew(NPAD * DIM_), 256, 0, stream>>>(pred, v_f, dpred_b, N_TOK * DIM_,
                                               NPAD * DIM_, c0);
    transpose_cvt<us><<<dim3(DIM_ / 32, NPAD / 32), tb, 0, stream>>>(dpred_b, dpredT,
                                                                     NPAD, DIM_);
    // grad_proj = h^T @ dpred : dpred column-major == dpredT
    gemm(stream, hT_b, dpredT, gproj, HID_, DIM_, NPAD, NPAD, NPAD, DIM_, HID_, nullptr);
    // dh = dpred @ proj^T : (proj^T) column-major == proj_tb
    gemm(stream, dpred_b, proj_tb, dhBuf, NPAD, HID_, DIM_, DIM_, DIM_, HID_, NPAD, nullptr);
    backact_k<<<ew(NPAD * HID_), 256, 0, stream>>>(gBuf, uBuf, dhBuf, dg_b, du_b,
                                                   NPAD * HID_);
    transpose_cvt<us><<<dim3(HID_ / 32, NPAD / 32), tb, 0, stream>>>(dg_b, dgT, NPAD, HID_);
    transpose_cvt<us><<<dim3(HID_ / 32, NPAD / 32), tb, 0, stream>>>(du_b, duT, NPAD, HID_);
    // grad_gate = k^T @ dg : dg column-major == dgT ; grad_val likewise
    gemm(stream, k_bT, dgT, ggate, DIM_, HID_, NPAD, NPAD, NPAD, HID_, DIM_, nullptr);
    gemm(stream, k_bT, duT, gval, DIM_, HID_, NPAD, NPAD, NPAD, HID_, DIM_, nullptr);
    ns_muon(gate, gate_b, ggate, DIM_, HID_, 0);
    transpose_cvt<float><<<dim3(HID_ / 32, DIM_ / 32), tb, 0, stream>>>(gate, gate_bT,
                                                                        DIM_, HID_);
    ns_muon(val, val_b, gval, DIM_, HID_, 0);
    transpose_cvt<float><<<dim3(HID_ / 32, DIM_ / 32), tb, 0, stream>>>(val, val_bT,
                                                                        DIM_, HID_);
    ns_muon(proj, proj_tb, gproj, HID_, DIM_, 1);   // refreshes proj_tb mirror
  }

  // ---- final forward with updated theta ----
  transpose_cvt<float><<<dim3(DIM_ / 32, HID_ / 32), tb, 0, stream>>>(proj, projT_b,
                                                                      HID_, DIM_);
  gemm(stream, q_b, gate_bT, gBuf, NPAD, HID_, DIM_, DIM_, DIM_, HID_, NPAD, nullptr);
  gemm(stream, q_b, val_bT, uBuf, NPAD, HID_, DIM_, DIM_, DIM_, HID_, NPAD, nullptr);
  swmul_k<<<ew(NPAD * HID_), 256, 0, stream>>>(gBuf, uBuf, h_b, NPAD * HID_);
  gemm(stream, h_b, projT_b, pred, NPAD, DIM_, HID_, HID_, HID_, DIM_, NPAD, nullptr);
  cvt_k<<<ew(NPAD * DIM_), 256, 0, stream>>>(pred, dpred_b, NPAD * DIM_, NPAD * DIM_);
  gemm(stream, dpred_b, projwT, out, NPAD, DIM_, DIM_, DIM_, DIM_, DIM_, N_TOK, proj_bias);
}
